// GlobalLinearCRFLayer_43997644980874
// MI455X (gfx1250) — compile-verified, matrix operand order checked by reading, not run on hardware
//
#include <hip/hip_runtime.h>
#include <hip/hip_bf16.h>

// ---------------------------------------------------------------------------
// Problem constants (from reference): B=128, S=512, H=1024, T_AUG=22
// ---------------------------------------------------------------------------
#define CRF_B      128
#define CRF_S      512
#define CRF_H      1024
#define CRF_T      22
#define CRF_START  20
#define CRF_END    21
#define CRF_NINF   (-10000.0f)
#define NT_PAD     32          // tags padded to 32 for WMMA / coalesced access
#define SENT       (-3.0e38f)  // sentinel for inactive lanes

typedef __attribute__((ext_vector_type(16))) __bf16 v16bf;
typedef __attribute__((ext_vector_type(8)))  float  v8f;

union BfU {
    __bf16       h[16];
    unsigned int u[8];
    v16bf        v;
};

// ---------------------------------------------------------------------------
// CDNA5 async global->LDS copy (GLOBAL_LOAD_ASYNC_TO_LDS_B128, ASYNCcnt).
// Probe result (round 2 diagnostic): the builtin exists and takes pointers to
// 128-bit int vectors: (v4i global*, v4i lds*, imm offset, imm cpol).
// ---------------------------------------------------------------------------
#if __has_builtin(__builtin_amdgcn_global_load_async_to_lds_b128)
#define HAVE_ASYNC_LDS 1
#else
#define HAVE_ASYNC_LDS 0
#endif

typedef int v4i __attribute__((vector_size(4 * sizeof(int))));
typedef __attribute__((address_space(1))) v4i* gv4i_p;   // global (device) AS
typedef __attribute__((address_space(3))) v4i* lv4i_p;   // LDS AS

__device__ __forceinline__ void copy16_to_lds(const float* g, float* l) {
#if HAVE_ASYNC_LDS
    // Generic->LDS: low 32 bits of a generic LDS pointer are the LDS offset.
    __builtin_amdgcn_global_load_async_to_lds_b128(
        (gv4i_p)(uintptr_t)g,
        (lv4i_p)(unsigned int)(uintptr_t)l,
        0, 0);
#else
    *reinterpret_cast<float4*>(l) = *reinterpret_cast<const float4*>(g);
#endif
}

template <int N>
__device__ __forceinline__ void wait_async_le() {
#if HAVE_ASYNC_LDS
#if __has_builtin(__builtin_amdgcn_s_wait_asynccnt)
    __builtin_amdgcn_s_wait_asynccnt(N);
#else
    asm volatile("s_wait_asynccnt %0" ::"n"(N));
#endif
#endif
}

// fp32 -> bf16 RNE pack used only in the tiny one-shot prep kernel
__device__ __forceinline__ unsigned int pack_bf16(float f0, float f1) {
    unsigned int u0 = __float_as_uint(f0);
    unsigned int u1 = __float_as_uint(f1);
    u0 = (u0 + 0x7FFFu + ((u0 >> 16) & 1u)) >> 16;
    u1 = (u1 + 0x7FFFu + ((u1 >> 16) & 1u)) >> 16;
    return (u1 << 16) | (u0 & 0xFFFFu);
}

// ---------------------------------------------------------------------------
// Kernel 1: build WMMA B-fragments of W (22x1024 fp32 -> bf16, padded to 32
// rows of zeros). Fragment layout mirrors the ISA 16-bit A layout:
//   lane L: N = L%16 ; K-half = (L<16 ? 0 : 8)
//   dword d in 0..3: K = ktile*32 + khalf + 2d, 2d+1
//   dword d in 4..7: K = ktile*32 + 16 + khalf + 2(d-4), ...
// Bfrag[(ktile*2 + ntile)*32 + lane][8 dwords]  -> 64 KB total, L2-resident.
// ---------------------------------------------------------------------------
__global__ __launch_bounds__(32)
void crf_prep_bfrag(const float* __restrict__ W, unsigned int* __restrict__ Bfrag)
{
    const int lane  = threadIdx.x;
    const int ntile = blockIdx.x & 1;
    const int ktile = blockIdx.x >> 1;
    const int n     = ntile * 16 + (lane & 15);
    const int khalf = (lane < 16) ? 0 : 8;

    unsigned int out[8];
#pragma unroll
    for (int d = 0; d < 8; ++d) {
        const int k = ktile * 32 + ((d < 4) ? 0 : 16) + khalf + 2 * (d & 3);
        const float f0 = (n < CRF_T) ? W[n * CRF_H + k]     : 0.0f;
        const float f1 = (n < CRF_T) ? W[n * CRF_H + k + 1] : 0.0f;
        out[d] = pack_bf16(f0, f1);
    }
    unsigned int* p = Bfrag + ((size_t)blockIdx.x * 32 + lane) * 8;
    *reinterpret_cast<uint4*>(p)     = *reinterpret_cast<uint4*>(&out[0]);
    *reinterpret_cast<uint4*>(p + 4) = *reinterpret_cast<uint4*>(&out[4]);
}

// ---------------------------------------------------------------------------
// Kernel 2: feats[M,32] = X[M,1024] * W^T (+ bias), M = B*S = 65536.
// HBM-bound: 256 MB fp32 activations @ 23.3 TB/s ~ 11 us. Double-buffered
// async global->LDS staging of 64x64 fp32 A tiles; bf16 WMMA for the math.
// Block = 128 threads (4 waves); wave computes 16 rows x 32 cols.
// ---------------------------------------------------------------------------
__global__ __launch_bounds__(128)
void crf_feat_gemm(const float* __restrict__ X,
                   const unsigned int* __restrict__ Bfrag,
                   const float* __restrict__ bias,
                   float* __restrict__ feats)
{
    __shared__ float As[2][64][68];   // 2 buffers, 64x64 fp32, +4 row padding

    const int    tid    = threadIdx.x;
    const int    lane   = tid & 31;
    const int    wave   = tid >> 5;
    const size_t rowBlk = (size_t)blockIdx.x * 64;

    const int khalf = (lane < 16) ? 0 : 8;   // ISA 16-bit A layout K-half
    const int arow  = wave * 16 + (lane & 15);

    v8f acc0 = {};  // columns 0..15
    v8f acc1 = {};  // columns 16..31 (22..31 are zero-padded W rows)

    // issue one 64x64 chunk: 1024 x b128, 8 per thread, coalesced per row
    auto issue_chunk = [&](int kc, float (*buf)[68]) {
#pragma unroll
        for (int j = 0; j < 8; ++j) {
            const int slot = tid + j * 128;      // 0..1023
            const int r    = slot >> 4;          // 16 x float4 per row
            const int c4   = slot & 15;
            copy16_to_lds(&X[(rowBlk + r) * CRF_H + kc + c4 * 4],
                          &buf[r][c4 * 4]);
        }
    };

    issue_chunk(0, As[0]);

    for (int c = 0; c < CRF_H / 64; ++c) {
        if (c + 1 < CRF_H / 64) {
            issue_chunk((c + 1) * 64, As[(c + 1) & 1]);
            wait_async_le<8>();   // the 8 just-issued may be outstanding
        } else {
            wait_async_le<0>();
        }
        __syncthreads();

        const float (*buf)[68] = As[c & 1];
        const int ktile0 = c * 2;

        // Hoist all B fragments for this chunk (L2-resident, pre-swizzled)
        BfU b00, b01, b10, b11;
        {
            const unsigned int* p;
            p = Bfrag + ((size_t)(ktile0 * 2 + 0) * 32 + lane) * 8;
            *reinterpret_cast<uint4*>(&b00.u[0]) = *reinterpret_cast<const uint4*>(p);
            *reinterpret_cast<uint4*>(&b00.u[4]) = *reinterpret_cast<const uint4*>(p + 4);
            p = Bfrag + ((size_t)(ktile0 * 2 + 1) * 32 + lane) * 8;
            *reinterpret_cast<uint4*>(&b01.u[0]) = *reinterpret_cast<const uint4*>(p);
            *reinterpret_cast<uint4*>(&b01.u[4]) = *reinterpret_cast<const uint4*>(p + 4);
            p = Bfrag + ((size_t)(ktile0 * 2 + 2) * 32 + lane) * 8;
            *reinterpret_cast<uint4*>(&b10.u[0]) = *reinterpret_cast<const uint4*>(p);
            *reinterpret_cast<uint4*>(&b10.u[4]) = *reinterpret_cast<const uint4*>(p + 4);
            p = Bfrag + ((size_t)(ktile0 * 2 + 3) * 32 + lane) * 8;
            *reinterpret_cast<uint4*>(&b11.u[0]) = *reinterpret_cast<const uint4*>(p);
            *reinterpret_cast<uint4*>(&b11.u[4]) = *reinterpret_cast<const uint4*>(p + 4);
        }

#pragma unroll
        for (int ks = 0; ks < 2; ++ks) {
            const int k0 = ks * 32;
            // A fragment (16x32 bf16): native f32->bf16 converts
            BfU a;
#pragma unroll
            for (int d = 0; d < 8; ++d) {
                const int col = (d < 4) ? (k0 + khalf + 2 * d)
                                        : (k0 + 16 + khalf + 2 * (d - 4));
                const float2 f = *reinterpret_cast<const float2*>(&buf[arow][col]);
                a.h[2 * d]     = (__bf16)f.x;
                a.h[2 * d + 1] = (__bf16)f.y;
            }
            acc0 = __builtin_amdgcn_wmma_f32_16x16x32_bf16(
                false, a.v, false, (ks ? b10 : b00).v, (short)0, acc0, false, false);
            acc1 = __builtin_amdgcn_wmma_f32_16x16x32_bf16(
                false, a.v, false, (ks ? b11 : b01).v, (short)0, acc1, false, false);
        }
        __syncthreads();   // before next iteration's async writes reuse buf
    }

    // C/D layout: VGPR j -> M = j + (lane<16 ? 0 : 8), N = lane%16
    const int   n0    = lane & 15;
    const int   n1    = n0 + 16;
    const int   mhi   = (lane >> 4) * 8;
    const float bias0 = bias[n0];
    const float bias1 = (n1 < CRF_T) ? bias[n1] : 0.0f;
    const size_t rbase = rowBlk + (size_t)wave * 16 + mhi;
#pragma unroll
    for (int j = 0; j < 8; ++j) {
        feats[(rbase + j) * NT_PAD + n0] = acc0[j] + bias0;
        feats[(rbase + j) * NT_PAD + n1] = acc1[j] + bias1;
    }
}

// ---------------------------------------------------------------------------
// Kernel 3: CRF forward scan + gold score, one wave per batch element.
// Reformulated recurrence:  m = max(alpha); e = exp(alpha - m);
//   alpha'[t] = feat[t] + m + log( sum_j exp(trans[t,j]) * e[j] )
// exp(trans) precomputed per-lane (22 regs); per step: 1 exp + 1 log +
// wave max + 22 FMA with readlane broadcasts. Loop runs only lengths[b]
// steps; the reference's i==len end-case collapses to adding trans[END,:]
// once after the loop.
// ---------------------------------------------------------------------------
__global__ __launch_bounds__(32)
void crf_scan_score(const float* __restrict__ feats,
                    const int* __restrict__ tags,
                    const int* __restrict__ lengths,
                    const float* __restrict__ trans,
                    float* __restrict__ out)
{
    __shared__ float sTrans[CRF_T * CRF_T];
    const int lane = threadIdx.x;
    const int b    = blockIdx.x;

    for (int i = lane; i < CRF_T * CRF_T; i += 32) sTrans[i] = trans[i];
    __syncthreads();

    // exp(trans[t, :]) for this lane's tag t (NINF entries underflow to 0)
    float expRow[CRF_T];
    {
        const int r = (lane < CRF_T) ? lane : 0;
#pragma unroll
        for (int j = 0; j < CRF_T; ++j) expRow[j] = __expf(sTrans[r * CRF_T + j]);
    }
    const float endAdd = (lane < CRF_T) ? sTrans[CRF_END * CRF_T + lane] : 0.0f;

    const int len = lengths[b];
    float alpha = (lane == CRF_START) ? 0.0f
                : ((lane < CRF_T) ? CRF_NINF : SENT);

    const float* fb = feats + (size_t)b * CRF_S * NT_PAD;
    for (int i = 0; i < len; ++i) {
        float m = alpha;
#pragma unroll
        for (int off = 16; off > 0; off >>= 1)
            m = fmaxf(m, __shfl_xor(m, off, 32));
        const float e = __expf(alpha - m);
        float s = 0.0f;
#pragma unroll
        for (int j = 0; j < CRF_T; ++j) {
            const float ej = __shfl(e, j, 32);     // constant lane -> readlane
            s = fmaf(expRow[j], ej, s);
        }
        const float feat = fb[(size_t)i * NT_PAD + lane];
        const float nv = feat + m + __logf(s);
        alpha = (lane < CRF_T) ? nv : SENT;
    }
    alpha += endAdd;

    // Z = logsumexp(alpha)
    float m = alpha;
#pragma unroll
    for (int off = 16; off > 0; off >>= 1)
        m = fmaxf(m, __shfl_xor(m, off, 32));
    float ssum = __expf(alpha - m);
#pragma unroll
    for (int off = 16; off > 0; off >>= 1)
        ssum += __shfl_xor(ssum, off, 32);
    const float Z = m + __logf(ssum);

    // gold score: sum over i<len of feats[i, tag_i] + trans[tag_i, tag_{i-1}]
    const int* tb = tags + (size_t)b * CRF_S;
    float partial = 0.0f;
    for (int i = lane; i < len; i += 32) {
        const int ti = tb[i];
        const int tp = (i == 0) ? CRF_START : tb[i - 1];
        partial += fb[(size_t)i * NT_PAD + ti] + sTrans[ti * CRF_T + tp];
    }
#pragma unroll
    for (int off = 16; off > 0; off >>= 1)
        partial += __shfl_xor(partial, off, 32);

    if (lane == 0) {
        const int lastTag = tb[len - 1];
        const float score = partial + sTrans[CRF_END * CRF_T + lastTag];
        out[b] = Z - score;
    }
}

// ---------------------------------------------------------------------------
// Launch. Workspace: [0, 64KB) pre-swizzled W fragments; then 8 MB feats.
// ---------------------------------------------------------------------------
extern "C" void kernel_launch(void* const* d_in, const int* in_sizes, int n_in,
                              void* d_out, int out_size, void* d_ws, size_t ws_size,
                              hipStream_t stream) {
    const float* X     = (const float*)d_in[0];   // [128,512,1024]
    const int*   tags  = (const int*)  d_in[1];   // [128,512]
    const int*   lens  = (const int*)  d_in[2];   // [128]
    const float* W     = (const float*)d_in[3];   // [22,1024]
    const float* bias  = (const float*)d_in[4];   // [22]
    const float* trans = (const float*)d_in[5];   // [22,22]
    float*       out   = (float*)d_out;           // [128]

    unsigned int* Bfrag = (unsigned int*)d_ws;                       // 64 KB
    float*        feats = (float*)((char*)d_ws + (64u << 10));       // 8 MB

    crf_prep_bfrag<<<64, 32, 0, stream>>>(W, Bfrag);
    crf_feat_gemm<<<(CRF_B * CRF_S) / 64, 128, 0, stream>>>(X, Bfrag, bias, feats);
    crf_scan_score<<<CRF_B, 32, 0, stream>>>(feats, tags, lens, trans, out);
}
